// RatioEstimatorCube_76802605187332
// MI455X (gfx1250) — matrix operand order repaired and checked by reading, not compiled, not run on hardware
//
#include <hip/hip_runtime.h>

// MI455X / gfx1250 fused histogram + suffix-sum (via WMMA) + masking kernel.
//
// Per batch b (4096 blocks, 256 threads = 8 wave32):
//   Phase 0: zero 64KB LDS histogram (ds_store_b128)
//   Phase 1: bin 1024 points with ds_add_f32 LDS atomics
//   Phase 2: suffix-sum along W as HIST(128x128) x T(128x128), T[k][j]=(k>=j),
//            using v_wmma_f32_16x16x32_f16 (hist values <=1024 are exact in f16,
//            accumulation in f32 -> bit-exact integer result)
//   Phase 3: stream counts -> global, and x * (counts>0) -> global (b128 ops)

typedef __attribute__((ext_vector_type(16))) _Float16 v16h;
typedef __attribute__((ext_vector_type(8)))  float    v8f;

#define NB    4096
#define HDIM  128
#define WDIM  128
#define HW    (HDIM * WDIM)     // 16384 floats = 64 KB LDS
#define NPTS  1024

__launch_bounds__(256)
__global__ void ratio_hist_fused(const float* __restrict__ x,
                                 const float* __restrict__ z,
                                 float* __restrict__ counts_out,
                                 float* __restrict__ masked_out)
{
    __shared__ float hist[HW];   // 64 KB of the 320 KB/WGP LDS

    const int tid = threadIdx.x;
    const int b   = blockIdx.x;

    // ---------------- Phase 0: zero LDS histogram (vectorized) ----------------
    {
        const float4 z4 = make_float4(0.f, 0.f, 0.f, 0.f);
        float4* h4 = (float4*)hist;
        #pragma unroll
        for (int v = 0; v < 16; ++v)
            h4[v * 256 + tid] = z4;          // ds_store_b128
    }
    __syncthreads();

    // ---------------- Phase 1: histogram (ds_add_f32 atomics) -----------------
    {
        // ranges are [0,1] -> bias=0, width=1 -> u == z
        const float2* zb = (const float2*)(z + (size_t)b * NPTS * 2);
        #pragma unroll
        for (int p = 0; p < 4; ++p) {
            const float2 pt = zb[p * 256 + tid];
            const bool in = (pt.x >= 0.f) && (pt.x <= 1.f) &&
                            (pt.y >= 0.f) && (pt.y <= 1.f);
            if (in) {
                const float scale = (float)HDIM * 0.9999999f;   // [0,dim)
                const int i0 = (int)(pt.x * scale);
                const int i1 = (int)(pt.y * scale);
                atomicAdd(&hist[i0 * WDIM + i1], 1.0f);         // ds_add_f32
            }
        }
    }
    __syncthreads();

    // ---------------- Phase 2: suffix-sum via WMMA ----------------------------
    // out = HIST x T,  T[k][j] = (k >= j) ? 1 : 0  (inclusive suffix sum)
    // Wave w owns output rows [16w, 16w+16); reads/writes only those LDS rows,
    // so no barrier is needed inside this phase.
    {
        const int wave   = tid >> 5;      // 0..7  == row tile ti
        const int lane   = tid & 31;
        const int laneHi = lane >> 4;     // 0/1
        const int laneLo = lane & 15;

        // Preload A fragments (rows 16*wave..+15, full K=128 as 4 chunks of 32).
        // 16-bit A 16x32 layout: lane = M + 16*laneHalf;
        //   vg 0..3 : K = 2*vg + {0,1} + 8*laneHi
        //   vg 4..7 : K = 16 + 2*(vg-4) + {0,1} + 8*laneHi
        v16h afrag[4];
        #pragma unroll
        for (int kc = 0; kc < 4; ++kc) {
            #pragma unroll
            for (int vg = 0; vg < 8; ++vg) {
                const int kbase = (vg < 4) ? 0 : 16;
                const int k0 = kbase + 2 * (vg & 3) + laneHi * 8;
                const float2 hv = *(const float2*)
                    &hist[(wave * 16 + laneLo) * WDIM + kc * 32 + k0];
                afrag[kc][2 * vg]     = (_Float16)hv.x;
                afrag[kc][2 * vg + 1] = (_Float16)hv.y;
            }
        }

        // Constant all-ones B fragment (for K-chunks entirely above diagonal).
        v16h vones;
        #pragma unroll
        for (int e = 0; e < 16; ++e) vones[e] = (_Float16)1.0f;

        #pragma unroll
        for (int tj = 0; tj < 8; ++tj) {
            v8f acc = {};
            #pragma unroll
            for (int kc = 0; kc < 4; ++kc) {
                if (kc * 32 + 31 < tj * 16) continue;       // B chunk all zero: skip
                if (kc * 32 >= tj * 16 + 16) {              // B chunk all ones
                    acc = __builtin_amdgcn_wmma_f32_16x16x32_f16(
                        false, afrag[kc], false, vones, (short)0, acc, false, false);
                } else {                                    // diagonal chunk: build B
                    // B 32x16 layout (matches A): vg 0..3 : K = 2*vg + {0,1} + 8*laneHi,
                    // vg 4..7 : K = 16 + 2*(vg-4) + {0,1} + 8*laneHi; N = lane%16.
                    v16h bf;
                    #pragma unroll
                    for (int e = 0; e < 16; ++e) {
                        const int vg = e >> 1;
                        const int k  = ((vg < 4) ? 0 : 16) + laneHi * 8 + 2 * (vg & 3) + (e & 1);
                        const int kg = kc * 32 + k;
                        const int ng = tj * 16 + laneLo;
                        bf[e] = (kg >= ng) ? (_Float16)1.0f : (_Float16)0.0f;
                    }
                    acc = __builtin_amdgcn_wmma_f32_16x16x32_f16(
                        false, afrag[kc], false, bf, (short)0, acc, false, false);
                }
            }
            // f32 C/D layout: VGPR r -> row r (lanes 0-15) / row r+8 (lanes 16-31),
            // col = lane%16. Overwrite hist in place (own rows only).
            #pragma unroll
            for (int r = 0; r < 8; ++r)
                hist[(wave * 16 + r + laneHi * 8) * WDIM + tj * 16 + laneLo] = acc[r];
        }
    }
    __syncthreads();

    // ---------------- Phase 3: stream counts + masked x (b128) ----------------
    {
        const float4* xb = (const float4*)(x + (size_t)b * HW);
        float4*       cb = (float4*)(counts_out + (size_t)b * HW);
        float4*       mb = (float4*)(masked_out + (size_t)b * HW);
        const float4* h4 = (const float4*)hist;
        #pragma unroll
        for (int v = 0; v < 16; ++v) {
            const int i4 = v * 256 + tid;
            const float4 c  = h4[i4];        // ds_load_b128
            const float4 xv = xb[i4];        // global_load_b128
            cb[i4] = c;                      // global_store_b128
            float4 o;
            o.x = (c.x > 0.f) ? xv.x : 0.f;
            o.y = (c.y > 0.f) ? xv.y : 0.f;
            o.z = (c.z > 0.f) ? xv.z : 0.f;
            o.w = (c.w > 0.f) ? xv.w : 0.f;
            mb[i4] = o;                      // global_store_b128
        }
    }
}

extern "C" void kernel_launch(void* const* d_in, const int* in_sizes, int n_in,
                              void* d_out, int out_size, void* d_ws, size_t ws_size,
                              hipStream_t stream) {
    (void)in_sizes; (void)n_in; (void)out_size; (void)d_ws; (void)ws_size;
    const float* x = (const float*)d_in[0];   // [4096,128,128] f32
    const float* z = (const float*)d_in[1];   // [4096,1024,2]  f32
    float* counts  = (float*)d_out;                              // output 0
    float* masked  = (float*)d_out + (size_t)NB * HW;            // output 1
    ratio_hist_fused<<<dim3(NB), dim3(256), 0, stream>>>(x, z, counts, masked);
}
